// GooseRecurrentLayer_37761352466560
// MI455X (gfx1250) — compile-verified, matrix operand order checked by reading, not run on hardware
//
#include <hip/hip_runtime.h>

#define D_MODEL 1024
#define SEQ     2048
#define BATCH   8
#define LN_EPS  1e-5f

typedef __attribute__((ext_vector_type(16))) _Float16 v16h;
typedef __attribute__((ext_vector_type(8)))  _Float16 v8h;
typedef __attribute__((ext_vector_type(8)))  float    v8f;

union Frag16 { v16h v; v8h h[2]; };

// ---------------------------------------------------------------------------
// Kernel 1: LayerNorm + token-shift mix -> f16 (xk, xv, xr)
// One block (256 threads, 8 waves) per token row; recompute LN stats of the
// previous token to avoid a second full pass over x.
// ---------------------------------------------------------------------------
__global__ __launch_bounds__(256)
void prep_kernel(const float* __restrict__ x,
                 const float* __restrict__ ln_w, const float* __restrict__ ln_b,
                 const float* __restrict__ tmk, const float* __restrict__ tmv,
                 const float* __restrict__ tmr,
                 _Float16* __restrict__ xk, _Float16* __restrict__ xv,
                 _Float16* __restrict__ xr)
{
    const int row = blockIdx.x;            // b*SEQ + t
    const int t   = row & (SEQ - 1);
    const bool has_prev = (t != 0);
    const float* xt = x + (size_t)row * D_MODEL;
    const float* xp = xt - D_MODEL;        // valid only if has_prev

    float s1 = 0.f, s2 = 0.f, p1 = 0.f, p2 = 0.f;
    for (int i = threadIdx.x; i < D_MODEL; i += 256) {
        float a = xt[i]; s1 += a; s2 += a * a;
        if (has_prev) { float b = xp[i]; p1 += b; p2 += b * b; }
    }
    #pragma unroll
    for (int off = 16; off > 0; off >>= 1) {
        s1 += __shfl_xor(s1, off);
        s2 += __shfl_xor(s2, off);
        p1 += __shfl_xor(p1, off);
        p2 += __shfl_xor(p2, off);
    }
    __shared__ float red[4][8];
    const int wid = threadIdx.x >> 5;
    if ((threadIdx.x & 31) == 0) {
        red[0][wid] = s1; red[1][wid] = s2; red[2][wid] = p1; red[3][wid] = p2;
    }
    __syncthreads();
    s1 = s2 = p1 = p2 = 0.f;
    #pragma unroll
    for (int w = 0; w < 8; ++w) {
        s1 += red[0][w]; s2 += red[1][w]; p1 += red[2][w]; p2 += red[3][w];
    }
    const float invD = 1.0f / (float)D_MODEL;
    const float m   = s1 * invD;
    const float var = s2 * invD - m * m;
    const float mp  = p1 * invD;
    const float vrp = p2 * invD - mp * mp;
    const float is  = rsqrtf(var + LN_EPS);
    const float isp = rsqrtf(vrp + LN_EPS);

    for (int i = threadIdx.x; i < D_MODEL; i += 256) {
        const float w  = ln_w[i], bb = ln_b[i];
        const float cur = (xt[i] - m) * is * w + bb;
        const float prv = has_prev ? ((xp[i] - mp) * isp * w + bb) : 0.f;
        const float mk = tmk[i], mv = tmv[i], mr = tmr[i];
        const size_t o = (size_t)row * D_MODEL + i;
        xk[o] = (_Float16)(cur * mk + prv * (1.f - mk));
        xv[o] = (_Float16)(cur * mv + prv * (1.f - mv));
        xr[o] = (_Float16)(cur * mr + prv * (1.f - mr));
    }
}

// ---------------------------------------------------------------------------
// Kernel 2: f32 -> f16 conversion (weights)
// ---------------------------------------------------------------------------
__global__ __launch_bounds__(256)
void cvt_f16(const float* __restrict__ s, _Float16* __restrict__ d, int n)
{
    int i = blockIdx.x * 256 + threadIdx.x;
    const int stride = gridDim.x * 256;
    for (; i < n; i += stride) d[i] = (_Float16)s[i];
}

// ---------------------------------------------------------------------------
// Kernel 3: WMMA GEMM  C[m,e] = sum_d A[m,d] * W[e,d]   (K = N = 1024)
// Block: 256 threads = 8 waves, 128x128 tile. Wave = 64(M) x 32(N) = 4x2
// wmma 16x16x32_f16 tiles. A/B fragments loaded directly from global in the
// ISA per-lane layout: lane half h = lane/16 owns K rows [8h..8h+7] (v0..3)
// and [16+8h..16+8h+7] (v4..7) -> two b128 loads per fragment per lane.
// mode 0: store acc; mode 1: store sigmoid(acc); mode 2: store X + R*acc.
// ---------------------------------------------------------------------------
__global__ __launch_bounds__(256)
void wmma_gemm_1024(const _Float16* __restrict__ A,   // M x 1024, row-major
                    const _Float16* __restrict__ W,   // 1024 x 1024, row-major (e,d)
                    float* __restrict__ Cout,         // M x 1024
                    const float* __restrict__ Xadd,   // mode 2 only
                    const float* __restrict__ Rmul,   // mode 2 only
                    int mode)
{
    constexpr int K = 1024, N = 1024;
    const int lane = threadIdx.x & 31;
    const int wave = threadIdx.x >> 5;
    const int wm = wave & 1;          // 0..1
    const int wn = wave >> 1;         // 0..3
    const int lh = lane >> 4;         // lane half: 0/1
    const int ln = lane & 15;

    const int m0 = blockIdx.x * 128 + wm * 64;   // 4 M-tiles of 16
    const int n0 = blockIdx.y * 128 + wn * 32;   // 2 N-tiles of 16

    v8f acc[4][2] = {};

    #pragma unroll 2
    for (int k0 = 0; k0 < K; k0 += 32) {
        Frag16 a[4], b[2];
        #pragma unroll
        for (int i = 0; i < 4; ++i) {
            const _Float16* p = A + (size_t)(m0 + i * 16 + ln) * K + k0 + 8 * lh;
            a[i].h[0] = *(const v8h*)p;
            a[i].h[1] = *(const v8h*)(p + 16);
        }
        #pragma unroll
        for (int j = 0; j < 2; ++j) {
            const _Float16* p = W + (size_t)(n0 + j * 16 + ln) * K + k0 + 8 * lh;
            b[j].h[0] = *(const v8h*)p;
            b[j].h[1] = *(const v8h*)(p + 16);
        }
        #pragma unroll
        for (int i = 0; i < 4; ++i)
            #pragma unroll
            for (int j = 0; j < 2; ++j)
                acc[i][j] = __builtin_amdgcn_wmma_f32_16x16x32_f16(
                    false, a[i].v, false, b[j].v, (short)0, acc[i][j], false, false);
    }

    // C layout: VGPR r -> row (r + 8*lh), col = lane%16
    #pragma unroll
    for (int i = 0; i < 4; ++i) {
        #pragma unroll
        for (int j = 0; j < 2; ++j) {
            #pragma unroll
            for (int r = 0; r < 8; ++r) {
                const int row = m0 + i * 16 + r + 8 * lh;
                const int col = n0 + j * 16 + ln;
                const size_t idx = (size_t)row * N + col;
                float val = acc[i][j][r];
                if (mode == 1)      val = 1.f / (1.f + __expf(-val));
                else if (mode == 2) val = Xadd[idx] + Rmul[idx] * val;
                Cout[idx] = val;
            }
        }
    }
}

// ---------------------------------------------------------------------------
// Kernel 4: per-channel scan  S_t = decay_d * S_{t-1} + k_t * v_t
// One thread per (b,d) channel (8192 threads), coalesced across d.
// Writes S_seq (f32, into d_out), final_S, and f16 S for the output GEMM.
// ---------------------------------------------------------------------------
__global__ __launch_bounds__(256)
void scan_kernel(const float* __restrict__ k, const float* __restrict__ v,
                 const float* __restrict__ decay_w,
                 float* __restrict__ s_seq, float* __restrict__ final_s,
                 _Float16* __restrict__ s16)
{
    const int ch = blockIdx.x * 256 + threadIdx.x;       // 0..8191
    const int b = ch / D_MODEL, d = ch % D_MODEL;
    const float dec = 1.f / (1.f + __expf(-decay_w[d]));
    const size_t base = (size_t)b * SEQ * D_MODEL + d;
    float S = 0.f;
    #pragma unroll 4
    for (int t = 0; t < SEQ; ++t) {
        const size_t idx = base + (size_t)t * D_MODEL;
        S = dec * S + k[idx] * v[idx];
        s_seq[idx] = S;
        s16[idx] = (_Float16)S;
    }
    final_s[ch] = S;
}

// ---------------------------------------------------------------------------
extern "C" void kernel_launch(void* const* d_in, const int* in_sizes, int n_in,
                              void* d_out, int out_size, void* d_ws, size_t ws_size,
                              hipStream_t stream)
{
    const float* x       = (const float*)d_in[0];
    const float* ln_w    = (const float*)d_in[1];
    const float* ln_b    = (const float*)d_in[2];
    const float* tmk     = (const float*)d_in[3];
    const float* tmv     = (const float*)d_in[4];
    const float* tmr     = (const float*)d_in[5];
    const float* decay_w = (const float*)d_in[6];
    const float* Wk      = (const float*)d_in[7];
    const float* Wv      = (const float*)d_in[8];
    const float* Wr      = (const float*)d_in[9];
    const float* Wo      = (const float*)d_in[10];

    constexpr size_t BT  = (size_t)BATCH * SEQ;       // 16384
    constexpr size_t BTD = BT * D_MODEL;              // 16,777,216
    constexpr size_t DD  = (size_t)D_MODEL * D_MODEL; // 1,048,576

    // workspace carve-out (~328 MB, all offsets 2 MB-aligned)
    char* ws = (char*)d_ws;
    const size_t need = BTD * 2 * 3 + DD * 2 * 4 + BTD * 4 * 3 + BTD * 2;
    if (ws_size < need) return;   // cannot run without scratch

    _Float16* xk16 = (_Float16*)ws; ws += BTD * 2;
    _Float16* xv16 = (_Float16*)ws; ws += BTD * 2;
    _Float16* xr16 = (_Float16*)ws; ws += BTD * 2;
    _Float16* Wk16 = (_Float16*)ws; ws += DD * 2;
    _Float16* Wv16 = (_Float16*)ws; ws += DD * 2;
    _Float16* Wr16 = (_Float16*)ws; ws += DD * 2;
    _Float16* Wo16 = (_Float16*)ws; ws += DD * 2;
    float* kbuf = (float*)ws; ws += BTD * 4;
    float* vbuf = (float*)ws; ws += BTD * 4;
    float* rbuf = (float*)ws; ws += BTD * 4;
    _Float16* S16 = (_Float16*)ws; ws += BTD * 2;

    // output layout: out (BTD), final_S (B*D), S_seq (BTD)
    float* out_p = (float*)d_out;
    float* finS  = out_p + BTD;
    float* Sseq  = finS + (size_t)BATCH * D_MODEL;

    // 1) LN + token-shift -> f16
    prep_kernel<<<(int)BT, 256, 0, stream>>>(x, ln_w, ln_b, tmk, tmv, tmr,
                                             xk16, xv16, xr16);
    // 2) weights -> f16
    cvt_f16<<<1024, 256, 0, stream>>>(Wk, Wk16, (int)DD);
    cvt_f16<<<1024, 256, 0, stream>>>(Wv, Wv16, (int)DD);
    cvt_f16<<<1024, 256, 0, stream>>>(Wr, Wr16, (int)DD);
    cvt_f16<<<1024, 256, 0, stream>>>(Wo, Wo16, (int)DD);

    // 3) k, v, r GEMMs
    dim3 g((unsigned)(BT / 128), D_MODEL / 128);  // (128, 8)
    wmma_gemm_1024<<<g, 256, 0, stream>>>(xk16, Wk16, kbuf, nullptr, nullptr, 0);
    wmma_gemm_1024<<<g, 256, 0, stream>>>(xv16, Wv16, vbuf, nullptr, nullptr, 0);
    wmma_gemm_1024<<<g, 256, 0, stream>>>(xr16, Wr16, rbuf, nullptr, nullptr, 1);

    // 4) recurrent scan (writes S_seq + final_S into d_out, f16 copy to ws)
    scan_kernel<<<(BATCH * D_MODEL) / 256, 256, 0, stream>>>(
        kbuf, vbuf, decay_w, Sseq, finS, S16);

    // 5) out = x + r * (S_seq @ Wo^T)
    wmma_gemm_1024<<<g, 256, 0, stream>>>(S16, Wo16, out_p, x, rbuf, 2);
}
